// Decoder_40879498729343
// MI455X (gfx1250) — matrix operand order repaired and checked by reading, not compiled, not run on hardware
//
#include <hip/hip_runtime.h>

// ---------------- problem constants (from reference) ----------------
#define VOCAB  512
#define NEMBD  256
#define LATENT 512
#define BATCH  512
#define MAXLEN 111

typedef __bf16 bf16_t;
typedef __attribute__((ext_vector_type(16))) __bf16 v16bf;
typedef __attribute__((ext_vector_type(8)))  __bf16 v8bf;
typedef __attribute__((ext_vector_type(8)))  float  v8f;

// ---------------- WMMA helpers (CDNA5 wave32, 16x16x32 bf16 -> f32) --------
__device__ __forceinline__ v8f wmma_bf16(v16bf a, v16bf b, v8f c) {
    // (neg_a, A, neg_b, B, c_mod, C, reuse_a, reuse_b)
    return __builtin_amdgcn_wmma_f32_16x16x32_bf16(false, a, false, b,
                                                   (short)0, c, false, false);
}

__device__ __forceinline__ v8bf ld8(const bf16_t* p) {
    return *(const v8bf*)p;   // 16-byte aligned by construction
}

__device__ __forceinline__ v16bf cat16(v8bf lo, v8bf hi) {
    v16bf o;
#pragma unroll
    for (int i = 0; i < 8; ++i) { o[i] = lo[i]; o[8 + i] = hi[i]; }
    return o;
}

// A fragment: 16x32 bf16 tile of row-major matrix.
// ISA layout: lanes 0-15 -> K[8g..8g+8) in V0-3, K[16+8g..) in V4-7, g=lane>>4.
__device__ __forceinline__ v16bf load_a(const bf16_t* rowbase, int g) {
    return cat16(ld8(rowbase + 8 * g), ld8(rowbase + 16 + 8 * g));
}
// B fragment for Bop[k,n] = W[n0+n, k] (W row-major): lane holds column n = lane&15,
// K[k0+8g..+8) then K[k0+16+8g..+8) of W row n0+n (same order as A).
__device__ __forceinline__ v16bf load_b(const bf16_t* wrow_k0, int g) {
    return cat16(ld8(wrow_k0 + 8 * g), ld8(wrow_k0 + 16 + 8 * g));
}

// ---------------- kernels ----------------

__global__ void cvt_f32_bf16(const float* __restrict__ src,
                             bf16_t* __restrict__ dst, int n) {
    int i = blockIdx.x * blockDim.x + threadIdx.x;
    if (i < n) dst[i] = (bf16_t)src[i];
}

// xs[t,b,h] = sum_e Wembd_bf[tok(b,t), e] * Wih_bf[h, e] + b_ih[h]
// 16(M) x 32(N) per wave: shared gathered-A, 2 B tiles. K = 256 -> 16 WMMA.
__global__ void xs_gemm(const bf16_t* __restrict__ Wembd,
                        const bf16_t* __restrict__ Wih,
                        const float*  __restrict__ bih,
                        const int*    __restrict__ teacher,
                        float* __restrict__ xs) {
    const int wid  = (blockIdx.x * blockDim.x + threadIdx.x) >> 5;
    const int lane = threadIdx.x & 31;
    const int g = lane >> 4, r = lane & 15;
    const int nj = wid & 15;            // 16 macro-cols of 32
    const int mi = (wid >> 4) & 31;     // 32 row tiles of 16
    const int t  = wid >> 9;
    if (t >= MAXLEN) return;
    const int m0 = mi * 16, n0 = nj * 32;

    const int brow = m0 + r;
    const int tok  = (t == 0) ? 1 : teacher[brow * MAXLEN + (t - 1)];
    const bf16_t* arow  = Wembd + (size_t)tok * NEMBD;
    const bf16_t* wrow0 = Wih + (size_t)(n0 + r) * NEMBD;
    const bf16_t* wrow1 = Wih + (size_t)(n0 + 16 + r) * NEMBD;

    v8f acc0 = {}, acc1 = {};
#pragma unroll
    for (int k0 = 0; k0 < NEMBD; k0 += 32) {
        const v16bf a = load_a(arow + k0, g);
        acc0 = wmma_bf16(a, load_b(wrow0 + k0, g), acc0);
        acc1 = wmma_bf16(a, load_b(wrow1 + k0, g), acc1);
    }

    const float bias0 = bih[n0 + r], bias1 = bih[n0 + 16 + r];
    float* ob = xs + (size_t)t * BATCH * LATENT;
#pragma unroll
    for (int d = 0; d < 8; ++d) {
        const size_t row = (size_t)(m0 + d + 8 * g) * LATENT;
        ob[row + n0 + r]      = acc0[d] + bias0;
        ob[row + n0 + 16 + r] = acc1[d] + bias1;
    }
}

// device-scope sense-reversing grid barrier
__device__ __forceinline__ void grid_barrier(unsigned* cnt, unsigned* gen, unsigned nb) {
    __threadfence();
    __syncthreads();
    if (threadIdx.x == 0) {
        unsigned g = __hip_atomic_load(gen, __ATOMIC_RELAXED, __HIP_MEMORY_SCOPE_AGENT);
        unsigned a = __hip_atomic_fetch_add(cnt, 1u, __ATOMIC_ACQ_REL, __HIP_MEMORY_SCOPE_AGENT);
        if (a == nb - 1u) {
            __hip_atomic_store(cnt, 0u, __ATOMIC_RELAXED, __HIP_MEMORY_SCOPE_AGENT);
            __hip_atomic_store(gen, g + 1u, __ATOMIC_RELEASE, __HIP_MEMORY_SCOPE_AGENT);
        } else {
            while (__hip_atomic_load(gen, __ATOMIC_ACQUIRE, __HIP_MEMORY_SCOPE_AGENT) == g)
                __builtin_amdgcn_s_sleep(2);
        }
    }
    __syncthreads();
}

// persistent recurrence: 32 WGs x 8 waves = 256 waves, one 32x32 macro-tile each
// (2x2 WMMA tiles, shared fragments -> 2 loads per WMMA, K = 512 -> 64 WMMA/step)
__global__ void rnn_persistent(const float*  __restrict__ xs,
                               const bf16_t* __restrict__ Whh,
                               const float*  __restrict__ bhh,
                               bf16_t* hA, bf16_t* hB,
                               bf16_t* __restrict__ hs,
                               unsigned* bar) {
    const int wid  = (blockIdx.x * blockDim.x + threadIdx.x) >> 5;   // 0..255
    const int lane = threadIdx.x & 31;
    const int g = lane >> 4, r = lane & 15;
    const unsigned nb = gridDim.x;

    const int mi = wid >> 4, ni = wid & 15;        // 16x16 macro grid of 32x32
    const int m0 = mi * 32, n0 = ni * 32;

    for (int t = 0; t < MAXLEN; ++t) {
        const bf16_t* hin  = (t & 1) ? hB : hA;
        bf16_t*       hout = (t & 1) ? hA : hB;

        // accumulators initialized with xs[t] tiles (f32)
        const float* xb = xs + ((size_t)t * BATCH + m0) * LATENT + n0;
        v8f acc00, acc01, acc10, acc11;
#pragma unroll
        for (int d = 0; d < 8; ++d) {
            const size_t r0 = (size_t)(d + 8 * g) * LATENT;
            const size_t r1 = (size_t)(16 + d + 8 * g) * LATENT;
            acc00[d] = xb[r0 + r];      acc01[d] = xb[r0 + 16 + r];
            acc10[d] = xb[r1 + r];      acc11[d] = xb[r1 + 16 + r];
        }

        const bf16_t* arow0 = hin + (size_t)(m0 + r) * LATENT;
        const bf16_t* arow1 = hin + (size_t)(m0 + 16 + r) * LATENT;
        const bf16_t* wrow0 = Whh + (size_t)(n0 + r) * LATENT;
        const bf16_t* wrow1 = Whh + (size_t)(n0 + 16 + r) * LATENT;
#pragma unroll
        for (int k0 = 0; k0 < LATENT; k0 += 32) {
            const v16bf a0 = load_a(arow0 + k0, g);
            const v16bf a1 = load_a(arow1 + k0, g);
            const v16bf b0 = load_b(wrow0 + k0, g);
            const v16bf b1 = load_b(wrow1 + k0, g);
            acc00 = wmma_bf16(a0, b0, acc00);
            acc01 = wmma_bf16(a0, b1, acc01);
            acc10 = wmma_bf16(a1, b0, acc10);
            acc11 = wmma_bf16(a1, b1, acc11);
        }

        const float bias0 = bhh[n0 + r], bias1 = bhh[n0 + 16 + r];
        bf16_t* ho = hout + (size_t)m0 * LATENT + n0;
        bf16_t* hb = hs + ((size_t)t * BATCH + m0) * LATENT + n0;
#pragma unroll
        for (int d = 0; d < 8; ++d) {
            const size_t r0 = (size_t)(d + 8 * g) * LATENT;
            const size_t r1 = (size_t)(16 + d + 8 * g) * LATENT;
            const bf16_t v00 = (bf16_t)tanhf(acc00[d] + bias0);
            const bf16_t v01 = (bf16_t)tanhf(acc01[d] + bias1);
            const bf16_t v10 = (bf16_t)tanhf(acc10[d] + bias0);
            const bf16_t v11 = (bf16_t)tanhf(acc11[d] + bias1);
            ho[r0 + r] = v00;      ho[r0 + 16 + r] = v01;
            ho[r1 + r] = v10;      ho[r1 + 16 + r] = v11;
            hb[r0 + r] = v00;      hb[r0 + 16 + r] = v01;
            hb[r1 + r] = v10;      hb[r1 + 16 + r] = v11;
        }

        grid_barrier(bar, bar + 1, nb);
    }
}

// out[b, t*V + v] = hs[t,b,:] . Wout[v,:] + b_out[v]  — 32x32 macro-tile per wave
__global__ void out_gemm(const bf16_t* __restrict__ hs,
                         const bf16_t* __restrict__ Wout,
                         const float*  __restrict__ bout,
                         float* __restrict__ out) {
    const int wid  = (blockIdx.x * blockDim.x + threadIdx.x) >> 5;
    const int lane = threadIdx.x & 31;
    const int g = lane >> 4, r = lane & 15;
    const int m  = wid & 255;            // 16x16 macro grid per step
    const int t  = wid >> 8;
    if (t >= MAXLEN) return;
    const int m0 = (m >> 4) * 32, n0 = (m & 15) * 32;

    const bf16_t* arow0 = hs + ((size_t)t * BATCH + m0 + r) * LATENT;
    const bf16_t* arow1 = hs + ((size_t)t * BATCH + m0 + 16 + r) * LATENT;
    const bf16_t* wrow0 = Wout + (size_t)(n0 + r) * LATENT;
    const bf16_t* wrow1 = Wout + (size_t)(n0 + 16 + r) * LATENT;

    v8f acc00 = {}, acc01 = {}, acc10 = {}, acc11 = {};
#pragma unroll
    for (int k0 = 0; k0 < LATENT; k0 += 32) {
        const v16bf a0 = load_a(arow0 + k0, g);
        const v16bf a1 = load_a(arow1 + k0, g);
        const v16bf b0 = load_b(wrow0 + k0, g);
        const v16bf b1 = load_b(wrow1 + k0, g);
        acc00 = wmma_bf16(a0, b0, acc00);
        acc01 = wmma_bf16(a0, b1, acc01);
        acc10 = wmma_bf16(a1, b0, acc10);
        acc11 = wmma_bf16(a1, b1, acc11);
    }

    const float bias0 = bout[n0 + r], bias1 = bout[n0 + 16 + r];
    const size_t ldo = (size_t)MAXLEN * VOCAB;
    float* ob = out + (size_t)t * VOCAB + n0;
#pragma unroll
    for (int d = 0; d < 8; ++d) {
        const size_t r0 = (size_t)(m0 + d + 8 * g) * ldo;
        const size_t r1 = (size_t)(m0 + 16 + d + 8 * g) * ldo;
        ob[r0 + r]      = acc00[d] + bias0;
        ob[r0 + 16 + r] = acc01[d] + bias1;
        ob[r1 + r]      = acc10[d] + bias0;
        ob[r1 + 16 + r] = acc11[d] + bias1;
    }
}

// ---------------- host launch ----------------
extern "C" void kernel_launch(void* const* d_in, const int* in_sizes, int n_in,
                              void* d_out, int out_size, void* d_ws, size_t ws_size,
                              hipStream_t stream) {
    const float* context = (const float*)d_in[1];
    const int*   teacher = (const int*)  d_in[2];
    const float* Wembd   = (const float*)d_in[3];
    const float* Wih     = (const float*)d_in[4];
    const float* Whh     = (const float*)d_in[5];
    const float* bih     = (const float*)d_in[6];
    const float* bhh     = (const float*)d_in[7];
    const float* Wout    = (const float*)d_in[8];
    const float* bout    = (const float*)d_in[9];
    float* out = (float*)d_out;

    // workspace layout (256B aligned)
    char* ws = (char*)d_ws;
    size_t off = 0;
    auto carve = [&](size_t bytes) { char* p = ws + off; off += (bytes + 255) & ~(size_t)255; return p; };
    unsigned* bar      = (unsigned*)carve(256);
    bf16_t* wembd_bf   = (bf16_t*)carve((size_t)VOCAB * NEMBD * 2);
    bf16_t* wih_bf     = (bf16_t*)carve((size_t)LATENT * NEMBD * 2);
    bf16_t* whh_bf     = (bf16_t*)carve((size_t)LATENT * LATENT * 2);
    bf16_t* wout_bf    = (bf16_t*)carve((size_t)VOCAB * LATENT * 2);
    bf16_t* hA         = (bf16_t*)carve((size_t)BATCH * LATENT * 2);
    bf16_t* hB         = (bf16_t*)carve((size_t)BATCH * LATENT * 2);
    bf16_t* hs         = (bf16_t*)carve((size_t)MAXLEN * BATCH * LATENT * 2);

    // xs (f32, L*B*H == out_size) staged in d_out — overwritten by out_gemm
    float* xs = out;

    // 1) bf16 weight conversion + h0 = context
    auto cvt = [&](const float* s, bf16_t* d, int n) {
        cvt_f32_bf16<<<(n + 255) / 256, 256, 0, stream>>>(s, d, n);
    };
    cvt(Wembd,   wembd_bf, VOCAB * NEMBD);
    cvt(Wih,     wih_bf,   LATENT * NEMBD);
    cvt(Whh,     whh_bf,   LATENT * LATENT);
    cvt(Wout,    wout_bf,  VOCAB * LATENT);
    cvt(context, hA,       BATCH * LATENT);

    // 2) xs = emb @ W_ih^T + b_ih : L * 32 * 16 waves of 16x32 tiles
    const int xs_waves = MAXLEN * (BATCH / 16) * (LATENT / 32);   // 56832
    xs_gemm<<<xs_waves / 8, 256, 0, stream>>>(wembd_bf, wih_bf, bih, teacher, xs);

    // 3) serial recurrence (persistent kernel, grid barrier in d_ws)
    hipMemsetAsync(bar, 0, 8, stream);
    rnn_persistent<<<32, 256, 0, stream>>>(xs, whh_bf, bhh, hA, hB, hs, bar);

    // 4) logits = hs @ W_out^T + b_out : L * 256 waves of 32x32 macro-tiles
    const int out_waves = MAXLEN * (BATCH / 32) * (VOCAB / 32) * 4; // 111*1024; only 111*256 do work
    out_gemm<<<out_waves / 8, 256, 0, stream>>>(hs, wout_bf, bout, out);
}
